// RPMultiHeadAttention_19043884990633
// MI455X (gfx1250) — compile-verified, matrix operand order checked by reading, not run on hardware
//
#include <hip/hip_runtime.h>
#include <hip/hip_bf16.h>

// ---------------------------------------------------------------------------
// RPMultiHeadAttention for MI455X (gfx1250, wave32, WMMA bf16 / f32-acc)
//   prep:  bf16 conversions of x / Wq / Wk / Wv / Wo ; cvec = context@Wc ;
//          p_bf16 = bf16(prev_state[...,:64])
//   gemm:  q/k/v (bf16 ws) = x_bf @ W^T  -- async-to-LDS double-buffered,
//          v_wmma_f32_16x16x32_bf16
//   dots:  qc/kc = per-head <q|c>, <k|c>
//   flash: fused scores = s/8 + a*qc*kc/64 - b*qp/8, mask, online softmax, PV
//   gemm:  out (f32) = attn_bf @ Wo^T
// ---------------------------------------------------------------------------

#define BB 2
#define SS 2048
#define DD 1024
#define HH_ 16
#define DH_ 64

typedef unsigned int u32;
typedef unsigned short u16;
typedef __attribute__((ext_vector_type(16))) __bf16 v16bf;
typedef __attribute__((ext_vector_type(8)))  float  v8f;
typedef __attribute__((ext_vector_type(8)))  u32    v8u;

union FragU { v8u u; v16bf b; };

__device__ __forceinline__ u16 f2bf(float f) {
  u32 u = __float_as_uint(f);
  u32 r = u + 0x7FFFu + ((u >> 16) & 1u);   // RNE
  return (u16)(r >> 16);
}
__device__ __forceinline__ float bflo(u32 u) { return __uint_as_float(u << 16); }
__device__ __forceinline__ float bfhi(u32 u) { return __uint_as_float(u & 0xFFFF0000u); }

// 16-bit A/B fragment K-index for (vgpr r, lane-half hh): ISA 7.12.2 layout.
__device__ __forceinline__ int fidx(int r, int hh) {
  return ((r >= 4) ? 16 : 0) + hh * 8 + ((r & 3) << 1);
}

// Generic pointer into LDS: hardware uses addr[31:0] as the LDS address.
__device__ __forceinline__ u32 lds_a(const void* p) { return (u32)(uintptr_t)p; }

// gfx1250 async global->LDS copies (ASYNCcnt); inline asm per CDNA5 ISA §10/§15.18.
__device__ __forceinline__ void async_b128(const void* g, u32 lds) {
  asm volatile("global_load_async_to_lds_b128 %0, %1, off"
               :: "v"(lds), "v"(g) : "memory");
}
__device__ __forceinline__ void async_b32(const void* g, u32 lds) {
  asm volatile("global_load_async_to_lds_b32 %0, %1, off"
               :: "v"(lds), "v"(g) : "memory");
}
__device__ __forceinline__ void wait_async0() {
  asm volatile("s_wait_asynccnt 0x0" ::: "memory");
}

__device__ __forceinline__ float rmax16(float v) {
#pragma unroll
  for (int m = 8; m >= 1; m >>= 1) v = fmaxf(v, __shfl_xor(v, m, 32));
  return v;
}
__device__ __forceinline__ float rsum16(float v) {
#pragma unroll
  for (int m = 8; m >= 1; m >>= 1) v += __shfl_xor(v, m, 32);
  return v;
}

// ---------------------------------------------------------------------------
// prep kernels
// ---------------------------------------------------------------------------
__global__ void cvt_bf16_kernel(const float* __restrict__ in,
                                u16* __restrict__ out, int n4) {
  int g = blockIdx.x * 256 + threadIdx.x;
  if (g < n4) {
    float4 v = ((const float4*)in)[g];
    ((u32*)out)[g * 2]     = f2bf(v.x) | ((u32)f2bf(v.y) << 16);
    ((u32*)out)[g * 2 + 1] = f2bf(v.z) | ((u32)f2bf(v.w) << 16);
  }
}

__global__ void ctx_proj_kernel(const float* __restrict__ context,
                                const float* __restrict__ Wc,
                                float* __restrict__ cvec) {
  int t = threadIdx.x;            // 128 threads: (b, f)
  int b = t >> 6, f = t & 63;
  float acc = 0.f;
  for (int d = 0; d < DD; ++d) acc += context[b * DD + d] * Wc[d * DH_ + f];
  cvec[t] = acc;
}

__global__ void prep_p_kernel(const float* __restrict__ prev,
                              u16* __restrict__ p_bf) {
  int g = blockIdx.x * 256 + threadIdx.x;     // B*S*64
  int row = g >> 6, f = g & 63;
  p_bf[g] = f2bf(prev[(size_t)row * DD + f]);
}

__global__ void ctx_dots_kernel(const u16* __restrict__ q_ws,
                                const u16* __restrict__ k_ws,
                                const float* __restrict__ cvec,
                                float* __restrict__ qc,
                                float* __restrict__ kc) {
  int g = blockIdx.x * 256 + threadIdx.x;     // over B*H*S
  const u16* src = blockIdx.y ? k_ws : q_ws;
  float* dst = blockIdx.y ? kc : qc;
  int b = g >> 15;                // H*S = 32768
  int rem = g & 32767;
  int h = rem >> 11;              // S = 2048
  int s = rem & 2047;
  const u32* s32 = (const u32*)src;
  size_t base = (((size_t)b * SS + s) * DD + h * DH_) >> 1;
  float acc = 0.f;
#pragma unroll 8
  for (int i = 0; i < 32; ++i) {
    u32 u = s32[base + i];
    acc += bflo(u) * cvec[b * DH_ + 2 * i] + bfhi(u) * cvec[b * DH_ + 2 * i + 1];
  }
  dst[g] = acc;                   // g == (b*H+h)*S + s
}

// ---------------------------------------------------------------------------
// WMMA GEMM: C[M,N] = A[M,K](bf16) @ W[N,K](bf16)^T
// block tile 128x64, BK=32, 8 waves; async-to-LDS double buffer
// ---------------------------------------------------------------------------
template <bool CF32>
__global__ __launch_bounds__(256) void gemm_wmma_kernel(
    const u16* __restrict__ A, const u16* __restrict__ Wbf,
    void* __restrict__ Cp, int M, int N, int K) {
  __shared__ u16 aLds[2][128 * 40];   // row stride 40 halves = 80 B (16B-aligned)
  __shared__ u16 bLds[2][64 * 40];

  const int tid = threadIdx.x;
  const int lane = tid & 31, w = tid >> 5;
  const int hh = lane >> 4, lm = lane & 15;
  const int waveM = w >> 1, waveN = w & 1;    // 4x2 wave grid
  const int rb0 = blockIdx.x * 128;
  const int nb0 = blockIdx.y * 64;

  auto fill = [&](int kb0, int buf) {
    // A tile 128x32: 128 rows * 4 x 16B = 512 transfers -> 2 per thread
#pragma unroll
    for (int i = 0; i < 2; ++i) {
      int e = tid + i * 256;
      int row = e >> 2, c = e & 3;
      async_b128(A + (size_t)(rb0 + row) * K + kb0 + c * 8,
                 lds_a(&aLds[buf][row * 40 + c * 8]));
    }
    // B tile 64x32: 256 transfers -> 1 per thread
    {
      int row = tid >> 2, c = tid & 3;
      async_b128(Wbf + (size_t)(nb0 + row) * K + kb0 + c * 8,
                 lds_a(&bLds[buf][row * 40 + c * 8]));
    }
  };

  v8f acc[2][2] = {};
  const int nk = K / 32;
  fill(0, 0);

  for (int ki = 0; ki < nk; ++ki) {
    const int cur = ki & 1;
    wait_async0();
    __syncthreads();
    if (ki + 1 < nk) fill((ki + 1) * 32, cur ^ 1);

    FragU a[2], bfr[2];
#pragma unroll
    for (int t = 0; t < 2; ++t)
#pragma unroll
      for (int r = 0; r < 8; ++r) {
        int k = fidx(r, hh);
        a[t].u[r]   = *(const u32*)&aLds[cur][(waveM * 32 + t * 16 + lm) * 40 + k];
        bfr[t].u[r] = *(const u32*)&bLds[cur][(waveN * 32 + t * 16 + lm) * 40 + k];
      }
#pragma unroll
    for (int i = 0; i < 2; ++i)
#pragma unroll
      for (int j = 0; j < 2; ++j)
        acc[i][j] = __builtin_amdgcn_wmma_f32_16x16x32_bf16(
            false, a[i].b, false, bfr[j].b, (short)0, acc[i][j], false, false);
  }

#pragma unroll
  for (int i = 0; i < 2; ++i)
#pragma unroll
    for (int j = 0; j < 2; ++j)
#pragma unroll
      for (int r = 0; r < 8; ++r) {
        int row = rb0 + waveM * 32 + i * 16 + r + 8 * hh;
        int col = nb0 + waveN * 32 + j * 16 + lm;
        if constexpr (CF32)
          ((float*)Cp)[(size_t)row * N + col] = acc[i][j][r];
        else
          ((u16*)Cp)[(size_t)row * N + col] = f2bf(acc[i][j][r]);
      }
}

// ---------------------------------------------------------------------------
// Fused flash attention with context-feature + prev-state terms
// grid: (B*H, S/128), block 256 (8 waves, 16 q-rows each)
// Double-buffered async staging of K/P/mask; V transposed via VALU path.
// ---------------------------------------------------------------------------
__global__ __launch_bounds__(256) void flash_attn_kernel(
    const u16* __restrict__ q_ws, const u16* __restrict__ k_ws,
    const u16* __restrict__ v_ws, const u16* __restrict__ p_bf,
    const float* __restrict__ qc, const float* __restrict__ kc,
    const unsigned char* __restrict__ mask,
    const float* __restrict__ alphaP, const float* __restrict__ betaP,
    u16* __restrict__ attn_ws) {
  __shared__ u16 kLds[2][32 * 72];       // [key][feat], stride 72 (144 B rows)
  __shared__ u16 pLds[2][32 * 72];
  __shared__ u16 vLds[2][64 * 40];       // transposed: [dh][key], stride 40
  __shared__ u32 mLds[2][128 * 8];       // 128 q-rows x 32 mask bytes
  __shared__ float kcLds[2][32];
  __shared__ u16 pstage[8][16 * 32];     // per-wave P re-layout (C -> A frag)

  const int tid = threadIdx.x;
  const int lane = tid & 31, w = tid >> 5;
  const int hh = lane >> 4, lm = lane & 15;
  const int bh = blockIdx.x;
  const int b = bh >> 4, h = bh & 15;
  const int qb0 = blockIdx.y * 128;
  const int qr0 = qb0 + w * 16;

  // Q fragments (16 rows x K=64 -> two K=32 A-fragments), straight from global
  const u32* q32 = (const u32*)q_ws;
  FragU qf[2];
  {
    size_t qb = ((size_t)b * SS + qr0 + lm) * DD + h * DH_;
#pragma unroll
    for (int c = 0; c < 2; ++c)
#pragma unroll
      for (int r = 0; r < 8; ++r)
        qf[c].u[r] = q32[(qb + c * 32 + fidx(r, hh)) >> 1];
  }
  float qcv[8];
#pragma unroll
  for (int r = 0; r < 8; ++r)
    qcv[r] = qc[(b * HH_ + h) * SS + qr0 + r + 8 * hh];

  const float alpha = alphaP[0] * (1.0f / 64.0f);
  const float beta  = betaP[0] * 0.125f;   // beta * inv_sqrt(DH)
  const float scl   = 0.125f;              // inv_sqrt(DH)

  const u32* v32 = (const u32*)v_ws;

  auto fill = [&](int kt, int buf) {
    const int key0 = kt * 32;
    // K & P tiles: 32 rows x 128 B each -> 256 B128 transfers apiece
    {
      int row = tid >> 3, c = tid & 7;
      async_b128(k_ws + ((size_t)b * SS + key0 + row) * DD + h * DH_ + c * 8,
                 lds_a(&kLds[buf][row * 72 + c * 8]));
      async_b128(p_bf + ((size_t)b * SS + key0 + row) * DH_ + c * 8,
                 lds_a(&pLds[buf][row * 72 + c * 8]));
    }
    // mask tile: 128 rows x 32 B = 1024 dwords
#pragma unroll
    for (int i = 0; i < 4; ++i) {
      int e = tid + i * 256;
      int mrow = e >> 3, mc = e & 7;
      async_b32(mask + ((size_t)b * SS + qb0 + mrow) * SS + key0 + mc * 4,
                lds_a(&mLds[buf][e]));
    }
    // V tile transposed into [dh][key] (needs VALU, stays on ds path)
#pragma unroll
    for (int i = 0; i < 4; ++i) {
      int e = tid + i * 256;                      // 1024 dwords
      int row = e >> 5, cd = e & 31;
      u32 uv = v32[((((size_t)b * SS + key0 + row) * DD + h * DH_) >> 1) + cd];
      vLds[buf][(2 * cd) * 40 + row]     = (u16)(uv & 0xFFFFu);
      vLds[buf][(2 * cd + 1) * 40 + row] = (u16)(uv >> 16);
    }
    if (tid < 32) kcLds[buf][tid] = kc[(b * HH_ + h) * SS + key0 + tid];
  };

  float rmax[8], rsum[8];
  v8f O[4] = {};
#pragma unroll
  for (int r = 0; r < 8; ++r) { rmax[r] = -3.0e38f; rsum[r] = 0.f; }

  fill(0, 0);

  for (int kt = 0; kt < SS / 32; ++kt) {
    const int cur = kt & 1;
    wait_async0();
    __syncthreads();                 // fences DS fills + all waves' async done
    if (kt + 1 < SS / 32) fill(kt + 1, cur ^ 1);

    // ---- scores: QK^T and QP^T over two 16-key subtiles
    v8f sqk[2] = {}, sqp[2] = {};
#pragma unroll
    for (int t = 0; t < 2; ++t)
#pragma unroll
      for (int c = 0; c < 2; ++c) {
        FragU bk, bp;
#pragma unroll
        for (int r = 0; r < 8; ++r) {
          int idx = ((t * 16 + lm) * 72 + c * 32 + fidx(r, hh)) >> 1;
          bk.u[r] = ((const u32*)&kLds[cur][0])[idx];
          bp.u[r] = ((const u32*)&pLds[cur][0])[idx];
        }
        sqk[t] = __builtin_amdgcn_wmma_f32_16x16x32_bf16(
            false, qf[c].b, false, bk.b, (short)0, sqk[t], false, false);
        sqp[t] = __builtin_amdgcn_wmma_f32_16x16x32_bf16(
            false, qf[c].b, false, bp.b, (short)0, sqp[t], false, false);
      }

    // ---- combine terms + mask
    float s0[8], s1[8];
    const float kc0 = kcLds[cur][lm], kc1 = kcLds[cur][16 + lm];
#pragma unroll
    for (int r = 0; r < 8; ++r) {
      int m = r + 8 * hh;
      s0[r] = scl * sqk[0][r] - beta * sqp[0][r] + alpha * qcv[r] * kc0;
      s1[r] = scl * sqk[1][r] - beta * sqp[1][r] + alpha * qcv[r] * kc1;
      const unsigned char* mrow = (const unsigned char*)&mLds[cur][0] + (w * 16 + m) * 32;
      if (!mrow[lm])      s0[r] = -1.0e9f;
      if (!mrow[16 + lm]) s1[r] = -1.0e9f;
    }

    // ---- online softmax over the 32-key chunk
#pragma unroll
    for (int r = 0; r < 8; ++r) {
      float tm = rmax16(fmaxf(s0[r], s1[r]));
      float nm = fmaxf(rmax[r], tm);
      float corr = __expf(rmax[r] - nm);
      rmax[r] = nm;
      float p0 = __expf(s0[r] - nm);
      float p1 = __expf(s1[r] - nm);
      rsum[r] = rsum[r] * corr + rsum16(p0 + p1);
#pragma unroll
      for (int d = 0; d < 4; ++d) O[d][r] *= corr;
      int m = r + 8 * hh;
      pstage[w][m * 32 + lm]      = f2bf(p0);
      pstage[w][m * 32 + 16 + lm] = f2bf(p1);
    }
    asm volatile("s_wait_dscnt 0x0" ::: "memory");  // P visible for re-layout read

    // ---- P (C-layout) -> A fragment via LDS, then PV accumulate
    FragU pf;
#pragma unroll
    for (int r = 0; r < 8; ++r)
      pf.u[r] = ((const u32*)pstage[w])[(lm * 32 + fidx(r, hh)) >> 1];
#pragma unroll
    for (int d = 0; d < 4; ++d) {
      FragU bv;
#pragma unroll
      for (int r = 0; r < 8; ++r)
        bv.u[r] = ((const u32*)&vLds[cur][0])[((d * 16 + lm) * 40 + fidx(r, hh)) >> 1];
      O[d] = __builtin_amdgcn_wmma_f32_16x16x32_bf16(
          false, pf.b, false, bv.b, (short)0, O[d], false, false);
    }
  }

  // ---- normalize and store attention output (bf16 ws for the Wo GEMM)
#pragma unroll
  for (int r = 0; r < 8; ++r) {
    float inv = 1.0f / rsum[r];
    int m = r + 8 * hh;
    size_t base = ((size_t)b * SS + qr0 + m) * DD + h * DH_ + lm;
#pragma unroll
    for (int d = 0; d < 4; ++d)
      attn_ws[base + d * 16] = f2bf(O[d][r] * inv);
  }
}

// ---------------------------------------------------------------------------
extern "C" void kernel_launch(void* const* d_in, const int* in_sizes, int n_in,
                              void* d_out, int out_size, void* d_ws, size_t ws_size,
                              hipStream_t stream) {
  (void)in_sizes; (void)n_in; (void)out_size; (void)ws_size;
  const float* x        = (const float*)d_in[0];
  const float* context  = (const float*)d_in[1];
  const float* prev     = (const float*)d_in[2];
  const unsigned char* mask = (const unsigned char*)d_in[3];
  const float* Wq       = (const float*)d_in[4];
  const float* Wk       = (const float*)d_in[5];
  const float* Wv       = (const float*)d_in[6];
  const float* Wo       = (const float*)d_in[7];
  const float* Wc       = (const float*)d_in[8];
  const float* alpha    = (const float*)d_in[9];
  const float* beta     = (const float*)d_in[10];

  char* ws = (char*)d_ws;
  const size_t MB = 1024 * 1024;
  u16* x_bf   = (u16*)(ws + 0 * MB);      // 8 MB (B*S*D bf16)
  u16* wq_bf  = (u16*)(ws + 8 * MB);      // 2 MB each
  u16* wk_bf  = (u16*)(ws + 10 * MB);
  u16* wv_bf  = (u16*)(ws + 12 * MB);
  u16* wo_bf  = (u16*)(ws + 14 * MB);
  u16* q_ws   = (u16*)(ws + 16 * MB);     // 8 MB each
  u16* k_ws   = (u16*)(ws + 24 * MB);
  u16* v_ws   = (u16*)(ws + 32 * MB);
  u16* attn_ws= (u16*)(ws + 40 * MB);
  u16* p_bf   = (u16*)(ws + 48 * MB);     // 512 KB
  float* qc   = (float*)(ws + 48 * MB + 512 * 1024);   // 256 KB
  float* kc   = (float*)(ws + 48 * MB + 768 * 1024);   // 256 KB
  float* cvec = (float*)(ws + 49 * MB);                // 512 B

  const int M = BB * SS;   // 4096

  cvt_bf16_kernel<<<(M * DD / 4 + 255) / 256, 256, 0, stream>>>(x, x_bf, M * DD / 4);
  cvt_bf16_kernel<<<(DD * DD / 4 + 255) / 256, 256, 0, stream>>>(Wq, wq_bf, DD * DD / 4);
  cvt_bf16_kernel<<<(DD * DD / 4 + 255) / 256, 256, 0, stream>>>(Wk, wk_bf, DD * DD / 4);
  cvt_bf16_kernel<<<(DD * DD / 4 + 255) / 256, 256, 0, stream>>>(Wv, wv_bf, DD * DD / 4);
  cvt_bf16_kernel<<<(DD * DD / 4 + 255) / 256, 256, 0, stream>>>(Wo, wo_bf, DD * DD / 4);
  ctx_proj_kernel<<<1, 128, 0, stream>>>(context, Wc, cvec);
  prep_p_kernel<<<(BB * SS * DH_) / 256, 256, 0, stream>>>(prev, p_bf);

  dim3 gg(M / 128, DD / 64);   // 32 x 16
  gemm_wmma_kernel<false><<<gg, 256, 0, stream>>>(x_bf, wq_bf, q_ws, M, DD, DD);
  gemm_wmma_kernel<false><<<gg, 256, 0, stream>>>(x_bf, wk_bf, k_ws, M, DD, DD);
  gemm_wmma_kernel<false><<<gg, 256, 0, stream>>>(x_bf, wv_bf, v_ws, M, DD, DD);

  ctx_dots_kernel<<<dim3((BB * HH_ * SS) / 256, 2), 256, 0, stream>>>(
      q_ws, k_ws, cvec, qc, kc);

  flash_attn_kernel<<<dim3(BB * HH_, SS / 128), 256, 0, stream>>>(
      q_ws, k_ws, v_ws, p_bf, qc, kc, mask, alpha, beta, attn_ws);

  gemm_wmma_kernel<true><<<gg, 256, 0, stream>>>(attn_ws, wo_bf, d_out, M, DD, DD);
}
